// ResnetCnnsovnetDynamicRouting_867583394574
// MI455X (gfx1250) — matrix operand order repaired and checked by reading, not compiled, not run on hardware
//
#include <hip/hip_runtime.h>
#include <math.h>

typedef __attribute__((ext_vector_type(16))) _Float16 v16h;
typedef __attribute__((ext_vector_type(8)))  float    v8f;

// =====================================================================
// Implicit-GEMM convolution via V_WMMA_F32_16X16X32_F16 (wave32).
// GEMM view: M = Nimg*Ho*Wo (output positions), N = Cout, K = Cin*KW*KW.
// One wave computes a 16(M) x 16*NT(N) strip: the A fragment is gathered
// once per 32-wide K chunk and reused across NT WMMAs (NT = Cout/16).
// mode 0: y = relu(conv + bias)
// mode 1: y = (conv + bias) * (gamma/sqrt(1+1e-5)) + beta   (BatchNorm)
// =====================================================================
template <int KW, int NT>
__global__ __launch_bounds__(32) void conv_wmma_kernel(
    const float* __restrict__ X, const float* __restrict__ Wt,
    const float* __restrict__ bias, const float* __restrict__ gamma,
    const float* __restrict__ beta, float* __restrict__ Y,
    int Nimg, int Cin, int H, int W, int Cout, int stride,
    int Ho, int Wo, int mode)
{
  constexpr int KK2 = KW * KW;
  const int lane = threadIdx.x;          // 0..31 (wave32)
  const int half = lane >> 4;            // 0 / 1
  const int l15  = lane & 15;

  const int HoWo = Ho * Wo;
  const int M    = Nimg * HoWo;
  const int K    = Cin * KK2;

  const int m0 = blockIdx.x * 16;        // M tile origin (Cout handled in-wave)

  // Decode the A-row this lane gathers (row M = m0 + l15).
  const int  mrow   = m0 + l15;
  const bool mvalid = mrow < M;
  int img = 0, oy = 0, ox = 0;
  if (mvalid) {
    img = mrow / HoWo;
    int r = mrow % HoWo;
    oy = r / Wo; ox = r % Wo;
  }
  const float* __restrict__ xbase =
      X + (size_t)img * Cin * H * W + (size_t)(oy * stride) * W + (ox * stride);

  v8f acc[NT];
#pragma unroll
  for (int nt = 0; nt < NT; ++nt) acc[nt] = v8f{};

  for (int k0 = 0; k0 < K; k0 += 32) {
    // ---- A fragment: 16x32 f16, contiguous K order (k = 16*half + j), matching B ----
    v16h a;
#pragma unroll
    for (int j = 0; j < 16; ++j) {
      const int kk = k0 + 16 * half + j;
      float v = 0.f;
      if (mvalid && kk < K) {
        const int ci  = kk / KK2;              // compile-time divisor
        const int rem = kk % KK2;
        const int ky  = rem / KW, kx = rem % KW;
        v = xbase[((size_t)ci * H + ky) * W + kx];
      }
      a[j] = (_Float16)v;
    }
    // Prefetch next weight chunk into the cache hierarchy (global_prefetch_b8).
    if (k0 + 32 < K) __builtin_prefetch(Wt + (size_t)k0 + 32, 0, 3);

    // ---- NT x { B fragment load + WMMA } reusing the A fragment ----
#pragma unroll
    for (int nt = 0; nt < NT; ++nt) {
      const int co = nt * 16 + l15;            // B column for this lane
      const int kb = k0 + 16 * half;           // first of 16 consecutive K values
      v16h b;
      if (co < Cout && kb + 16 <= K) {
        const float* __restrict__ wp = Wt + (size_t)co * K + kb;
#pragma unroll
        for (int j = 0; j < 16; ++j) b[j] = (_Float16)wp[j];
      } else {
#pragma unroll
        for (int j = 0; j < 16; ++j) {
          const int kk = kb + j;
          float v = 0.f;
          if (co < Cout && kk < K) v = Wt[(size_t)co * K + kk];
          b[j] = (_Float16)v;
        }
      }
      acc[nt] = __builtin_amdgcn_wmma_f32_16x16x32_f16(
          /*neg_a=*/false, a, /*neg_b=*/false, b,
          /*c_mod=*/(short)0, acc[nt], /*reuse_a=*/false, /*reuse_b=*/false);
    }
  }

  // ---- epilogue: C layout VGPR r -> row (r + 8*half), col l15 ----
#pragma unroll
  for (int nt = 0; nt < NT; ++nt) {
    const int co = nt * 16 + l15;
    if (co >= Cout) continue;
    const float bi = bias[co];
    float gs = 0.f, bb = 0.f;
    if (mode == 1) {
      gs = gamma[co] * 0.999995000037499687f;   // 1/sqrt(1+1e-5)
      bb = beta[co];
    }
#pragma unroll
    for (int r = 0; r < 8; ++r) {
      const int mr = m0 + r + 8 * half;
      if (mr >= M) continue;
      float v = acc[nt][r] + bi;
      v = (mode == 0) ? fmaxf(v, 0.f) : (v * gs + bb);
      const int img2 = mr / HoWo;
      const int rr   = mr % HoWo;
      Y[((size_t)img2 * Cout + co) * HoWo + rr] = v;
    }
  }
}

template <int KW, int NT>
static void run_conv(const float* X, const float* Wt, const float* bias,
                     const float* g, const float* be, float* Y,
                     int Nimg, int Cin, int H, int W, int Cout, int s, int mode,
                     hipStream_t stream)
{
  const int Ho = (H - KW) / s + 1, Wo = (W - KW) / s + 1;
  const int M  = Nimg * Ho * Wo;
  dim3 grid((M + 15) / 16, 1);
  conv_wmma_kernel<KW, NT><<<grid, dim3(32), 0, stream>>>(
      X, Wt, bias, g, be, Y, Nimg, Cin, H, W, Cout, s, Ho, Wo, mode);
}

// ============ LayerNorm over (C,H,W) per sample — 3-kernel scheme ============
static constexpr int LN_CH = 64;   // reduction chunks per sample

__global__ __launch_bounds__(256) void ln_partial_kernel(
    const float* __restrict__ x, float* __restrict__ part, int n)
{
  const int s = blockIdx.x, c = blockIdx.y;
  const int len = (n + LN_CH - 1) / LN_CH;
  const int lo = c * len, hi = min(n, lo + len);
  const float* xs = x + (size_t)s * n;
  __shared__ float r1[256], r2[256];
  float sum = 0.f, sq = 0.f;
  for (int i = lo + threadIdx.x; i < hi; i += 256) {
    float v = xs[i]; sum += v; sq += v * v;
  }
  r1[threadIdx.x] = sum; r2[threadIdx.x] = sq;
  __syncthreads();
  for (int ofs = 128; ofs > 0; ofs >>= 1) {
    if (threadIdx.x < ofs) { r1[threadIdx.x] += r1[threadIdx.x + ofs];
                             r2[threadIdx.x] += r2[threadIdx.x + ofs]; }
    __syncthreads();
  }
  if (threadIdx.x == 0) {
    part[(s * LN_CH + c) * 2 + 0] = r1[0];
    part[(s * LN_CH + c) * 2 + 1] = r2[0];
  }
}

__global__ __launch_bounds__(64) void ln_final_kernel(
    const float* __restrict__ part, float* __restrict__ stats, int n)
{
  const int s = blockIdx.x;
  __shared__ float r1[64], r2[64];
  r1[threadIdx.x] = part[(s * LN_CH + threadIdx.x) * 2 + 0];
  r2[threadIdx.x] = part[(s * LN_CH + threadIdx.x) * 2 + 1];
  __syncthreads();
  for (int ofs = 32; ofs > 0; ofs >>= 1) {
    if (threadIdx.x < ofs) { r1[threadIdx.x] += r1[threadIdx.x + ofs];
                             r2[threadIdx.x] += r2[threadIdx.x + ofs]; }
    __syncthreads();
  }
  if (threadIdx.x == 0) {
    const float mean = r1[0] / (float)n;
    const float var  = r2[0] / (float)n - mean * mean;
    stats[s * 2 + 0] = mean;
    stats[s * 2 + 1] = rsqrtf(var + 1e-5f);
  }
}

__global__ void ln_apply_kernel(
    const float* __restrict__ x, const float* __restrict__ stats,
    const float* __restrict__ g, const float* __restrict__ b,
    float* __restrict__ y, int n)
{
  const int idx = blockIdx.x * blockDim.x + threadIdx.x;   // over 8*n
  if (idx >= 8 * n) return;
  const int s = idx / n, i = idx % n;
  y[idx] = (x[idx] - stats[s * 2]) * stats[s * 2 + 1] * g[i] + b[i];
}

// ============ squash primary caps: (8,8,16,HW), norm over the 16-dim ============
__global__ void squash_primary_kernel(const float* __restrict__ p,
                                      float* __restrict__ out, int HW)
{
  const int idx = blockIdx.x * blockDim.x + threadIdx.x;   // over 8*8*HW
  if (idx >= 8 * 8 * HW) return;
  const int pix = idx % HW;
  const int bi  = idx / HW;                                // b*8+i
  const float* base = p + ((size_t)bi * 16) * HW + pix;
  float n2 = 0.f;
#pragma unroll
  for (int d = 0; d < 16; ++d) { float v = base[(size_t)d * HW]; n2 += v * v; }
  const float n  = sqrtf(n2);
  const float sc = n2 / (1.f + n2) / (n + 1e-8f);
  float* ob = out + ((size_t)bi * 16) * HW + pix;
#pragma unroll
  for (int d = 0; d < 16; ++d) ob[(size_t)d * HW] = sc * base[(size_t)d * HW];
}

// ==== capsule combine: p(64, outCaps*16, HW) -> v(8, outCaps, 16, HW) ====
__global__ void caps_combine_kernel(const float* __restrict__ p,
                                    float* __restrict__ v, int outCaps, int HW)
{
  const int idx = blockIdx.x * blockDim.x + threadIdx.x;   // over 8*outCaps*HW
  if (idx >= 8 * outCaps * HW) return;
  const int pix = idx % HW;
  const int t   = idx / HW;
  const int oc  = t % outCaps;
  const int b   = t / outCaps;
  float s[16];
#pragma unroll
  for (int d = 0; d < 16; ++d) s[d] = 0.f;
  for (int i = 0; i < 8; ++i) {
    const float* base = p + (((size_t)(b * 8 + i) * outCaps + oc) * 16) * HW + pix;
#pragma unroll
    for (int d = 0; d < 16; ++d) s[d] += base[(size_t)d * HW];
  }
  const float cs = 1.f / (float)outCaps;
  float n2 = 0.f;
#pragma unroll
  for (int d = 0; d < 16; ++d) { s[d] *= cs; n2 += s[d] * s[d]; }
  const float n  = sqrtf(n2);
  const float sc = n2 / (1.f + n2) / (n + 1e-8f);
  float* ob = v + (((size_t)b * outCaps + oc) * 16) * HW + pix;
#pragma unroll
  for (int d = 0; d < 16; ++d) ob[(size_t)d * HW] = sc * s[d];
}

// ==== head: vc(8,2,16) -> class_pre(8,2) in out[0..15]; S = sum_b p0[b] ====
__global__ __launch_bounds__(32) void head_kernel(const float* __restrict__ vc,
                                                  float* __restrict__ out,
                                                  float* __restrict__ Sout)
{
  __shared__ float nrm[8][2];
  const int t = threadIdx.x;
  if (t < 16) {
    const int b = t >> 1, c = t & 1;
    const float* p = vc + (b * 2 + c) * 16;
    float s = 0.f;
#pragma unroll
    for (int d = 0; d < 16; ++d) s += p[d] * p[d];
    nrm[b][c] = sqrtf(s);
  }
  __syncthreads();
  if (t < 8) {
    const float a = nrm[t][0], c1 = nrm[t][1];
    const float mx = fmaxf(a, c1);
    const float e0 = expf(a - mx), e1 = expf(c1 - mx);
    const float z = e0 + e1;
    out[t * 2 + 0] = e0 / z;
    out[t * 2 + 1] = e1 / z;
    nrm[t][0] = e0 / z;                     // p0[b]
  }
  __syncthreads();
  if (t == 0) {
    float S = 0.f;
    for (int b = 0; b < 8; ++b) S += nrm[b][0];
    Sout[0] = S;
  }
}

// ==== t4 = 0.5 * ||v4[b,o,:,pix]|| * S   (cross-batch map_out collapsed) ====
__global__ void t4_kernel(const float* __restrict__ v4,
                          const float* __restrict__ S, float* __restrict__ t4)
{
  const int idx = blockIdx.x * blockDim.x + threadIdx.x;   // 8*8*16
  if (idx >= 8 * 8 * 16) return;
  const int pix = idx % 16;
  const int bo  = idx / 16;
  const float* vb = v4 + ((size_t)bo * 16) * 16 + pix;
  float n2 = 0.f;
#pragma unroll
  for (int d = 0; d < 16; ++d) { float v = vb[d * 16]; n2 += v * v; }
  t4[idx] = 0.5f * S[0] * sqrtf(n2);
}

// ==== all-ones conv_transpose, summed over the 8 input channels ====
// in (8,8,Hi,Wi) -> out (8,Ho,Wo);  Ho=(Hi-1)*s+k
__global__ void convt_kernel(const float* __restrict__ in, float* __restrict__ out,
                             int Hi, int Wi, int k, int s, int Ho, int Wo)
{
  const int idx = blockIdx.x * blockDim.x + threadIdx.x;   // 8*Ho*Wo
  if (idx >= 8 * Ho * Wo) return;
  const int x = idx % Wo;
  const int y = (idx / Wo) % Ho;
  const int b = idx / (Wo * Ho);
  const int hlo = (y >= k) ? ((y - k) / s + 1) : 0;
  const int hhi = min(Hi - 1, y / s);
  const int wlo = (x >= k) ? ((x - k) / s + 1) : 0;
  const int whi = min(Wi - 1, x / s);
  float sum = 0.f;
  for (int c = 0; c < 8; ++c) {
    const float* ib = in + (size_t)(b * 8 + c) * Hi * Wi;
    for (int h = hlo; h <= hhi; ++h)
      for (int w = wlo; w <= whi; ++w) sum += ib[h * Wi + w];
  }
  out[idx] = sum;
}

// ==== t = ||v[b,o,:,pix]|| * m[b,pix]  (cij=1/8 sums collapse to 1) ====
__global__ void t_mul_kernel(const float* __restrict__ v, const float* __restrict__ m,
                             float* __restrict__ t, int HW)
{
  const int idx = blockIdx.x * blockDim.x + threadIdx.x;   // 8*8*HW
  if (idx >= 8 * 8 * HW) return;
  const int pix = idx % HW;
  const int bo  = idx / HW;
  const int b   = bo / 8;
  const float* vb = v + ((size_t)bo * 16) * HW + pix;
  float n2 = 0.f;
#pragma unroll
  for (int d = 0; d < 16; ++d) { float x = vb[(size_t)d * HW]; n2 += x * x; }
  t[idx] = sqrtf(n2) * m[(size_t)b * HW + pix];
}

// ==== map_caps[b,d,pix] = ||primary[b,:,d,pix]|| * 8 * msum1[b,pix] ====
__global__ void map_caps_kernel(const float* __restrict__ primary,
                                const float* __restrict__ m1, float* __restrict__ out)
{
  const int idx = blockIdx.x * blockDim.x + threadIdx.x;   // 8*16*900
  if (idx >= 8 * 16 * 900) return;
  const int pix = idx % 900;
  const int d   = (idx / 900) % 16;
  const int b   = idx / (900 * 16);
  float n2 = 0.f;
#pragma unroll
  for (int i = 0; i < 8; ++i) {
    float v = primary[(((size_t)b * 8 + i) * 16 + d) * 900 + pix];
    n2 += v * v;
  }
  out[idx] = sqrtf(n2) * 8.f * m1[(size_t)b * 900 + pix];
}

// =====================================================================
// Workspace layout (floats)
// =====================================================================
static constexpr size_t O_A  = 0;                         // conv1/ln1 (8,32,126,126)
static constexpr size_t SZ_A = 8ull * 32 * 126 * 126;
static constexpr size_t O_B  = O_A + SZ_A;                // conv2/ln2 (8,32,62,62)
static constexpr size_t SZ_B = 8ull * 32 * 62 * 62;
static constexpr size_t O_C  = O_B + SZ_B;                // pc out (8,128,30,30)
static constexpr size_t SZ_C = 8ull * 128 * 30 * 30;
static constexpr size_t O_P  = O_C + SZ_C;                // primary squashed
static constexpr size_t O_PC = O_P + SZ_C;                // caps conv scratch (max 64,128,28,28)
static constexpr size_t SZ_PC = 64ull * 128 * 28 * 28;
static constexpr size_t O_V1 = O_PC + SZ_PC;              // (8,8,16,784)
static constexpr size_t O_V2 = O_V1 + 8ull * 8 * 16 * 784;
static constexpr size_t O_V3 = O_V2 + 8ull * 8 * 16 * 169;
static constexpr size_t O_V4 = O_V3 + 8ull * 8 * 16 * 81;
static constexpr size_t O_VC = O_V4 + 8ull * 8 * 16 * 16; // (8,2,16)
static constexpr size_t O_T4 = O_VC + 256;
static constexpr size_t O_M4 = O_T4 + 1024;               // (8,9,9)
static constexpr size_t O_T3 = O_M4 + 8ull * 81;
static constexpr size_t O_M3 = O_T3 + 8ull * 8 * 81;      // (8,13,13)
static constexpr size_t O_T2 = O_M3 + 8ull * 169;
static constexpr size_t O_M2 = O_T2 + 8ull * 8 * 169;     // (8,28,28)
static constexpr size_t O_T1 = O_M2 + 8ull * 784;
static constexpr size_t O_M1 = O_T1 + 8ull * 8 * 784;     // (8,30,30)
static constexpr size_t O_S  = O_M1 + 8ull * 900;         // scalar S
static constexpr size_t O_PART = O_S + 8;                 // LN partials 8*64*2
static constexpr size_t O_ST   = O_PART + 8ull * LN_CH * 2; // LN stats 8*2

extern "C" void kernel_launch(void* const* d_in, const int* in_sizes, int n_in,
                              void* d_out, int out_size, void* d_ws, size_t ws_size,
                              hipStream_t stream)
{
  const float* x     = (const float*)d_in[0];
  const float* c1_w  = (const float*)d_in[1];
  const float* c1_b  = (const float*)d_in[2];
  const float* ln1_g = (const float*)d_in[3];
  const float* ln1_b = (const float*)d_in[4];
  const float* c2_w  = (const float*)d_in[5];
  const float* c2_b  = (const float*)d_in[6];
  const float* ln2_g = (const float*)d_in[7];
  const float* ln2_b = (const float*)d_in[8];
  const float* pc_w  = (const float*)d_in[9];
  const float* pc_b  = (const float*)d_in[10];
  const float* pc_g  = (const float*)d_in[11];
  const float* pc_be = (const float*)d_in[12];
  const float *cw[5], *cb[5], *cg[5], *cbe[5];
  for (int i = 0; i < 5; ++i) {
    cw[i]  = (const float*)d_in[13 + 4 * i + 0];
    cb[i]  = (const float*)d_in[13 + 4 * i + 1];
    cg[i]  = (const float*)d_in[13 + 4 * i + 2];
    cbe[i] = (const float*)d_in[13 + 4 * i + 3];
  }
  float* ws  = (float*)d_ws;
  float* out = (float*)d_out;

  auto ew = [&](int total) { return dim3((total + 255) / 256); };
  auto ln = [&](float* buf, const float* g, const float* b, int n) {
    ln_partial_kernel<<<dim3(8, LN_CH), 256, 0, stream>>>(buf, ws + O_PART, n);
    ln_final_kernel<<<8, 64, 0, stream>>>(ws + O_PART, ws + O_ST, n);
    ln_apply_kernel<<<ew(8 * n), 256, 0, stream>>>(buf, ws + O_ST, g, b, buf, n);
  };

  // ---- stem ----
  run_conv<3, 2>(x,        c1_w, c1_b, nullptr, nullptr, ws + O_A, 8, 3, 128, 128, 32, 1, 0, stream);
  ln(ws + O_A, ln1_g, ln1_b, 32 * 126 * 126);
  run_conv<3, 2>(ws + O_A, c2_w, c2_b, nullptr, nullptr, ws + O_B, 8, 32, 126, 126, 32, 2, 0, stream);
  ln(ws + O_B, ln2_g, ln2_b, 32 * 62 * 62);

  // ---- primary caps ----
  run_conv<3, 8>(ws + O_B, pc_w, pc_b, pc_g, pc_be, ws + O_C, 8, 32, 62, 62, 128, 2, 1, stream);
  squash_primary_kernel<<<ew(8 * 8 * 900), 256, 0, stream>>>(ws + O_C, ws + O_P, 900);

  // ---- conv capsule stages ----
  run_conv<3, 8>(ws + O_P,  cw[0], cb[0], cg[0], cbe[0], ws + O_PC, 64, 16, 30, 30, 128, 1, 1, stream);
  caps_combine_kernel<<<ew(8 * 8 * 784), 256, 0, stream>>>(ws + O_PC, ws + O_V1, 8, 784);
  run_conv<3, 8>(ws + O_V1, cw[1], cb[1], cg[1], cbe[1], ws + O_PC, 64, 16, 28, 28, 128, 2, 1, stream);
  caps_combine_kernel<<<ew(8 * 8 * 169), 256, 0, stream>>>(ws + O_PC, ws + O_V2, 8, 169);
  run_conv<5, 8>(ws + O_V2, cw[2], cb[2], cg[2], cbe[2], ws + O_PC, 64, 16, 13, 13, 128, 1, 1, stream);
  caps_combine_kernel<<<ew(8 * 8 * 81), 256, 0, stream>>>(ws + O_PC, ws + O_V3, 8, 81);
  run_conv<3, 8>(ws + O_V3, cw[3], cb[3], cg[3], cbe[3], ws + O_PC, 64, 16, 9, 9, 128, 2, 1, stream);
  caps_combine_kernel<<<ew(8 * 8 * 16), 256, 0, stream>>>(ws + O_PC, ws + O_V4, 8, 16);
  run_conv<4, 2>(ws + O_V4, cw[4], cb[4], cg[4], cbe[4], ws + O_PC, 64, 16, 4, 4, 32, 1, 1, stream);
  caps_combine_kernel<<<ew(8 * 2 * 1), 256, 0, stream>>>(ws + O_PC, ws + O_VC, 2, 1);

  // ---- head + reconstruction tail ----
  head_kernel<<<1, 32, 0, stream>>>(ws + O_VC, out, ws + O_S);
  t4_kernel<<<ew(1024), 256, 0, stream>>>(ws + O_V4, ws + O_S, ws + O_T4);
  convt_kernel<<<ew(8 * 81), 256, 0, stream>>>(ws + O_T4, ws + O_M4, 4, 4, 3, 2, 9, 9);
  t_mul_kernel<<<ew(8 * 8 * 81), 256, 0, stream>>>(ws + O_V3, ws + O_M4, ws + O_T3, 81);
  convt_kernel<<<ew(8 * 169), 256, 0, stream>>>(ws + O_T3, ws + O_M3, 9, 9, 5, 1, 13, 13);
  t_mul_kernel<<<ew(8 * 8 * 169), 256, 0, stream>>>(ws + O_V2, ws + O_M3, ws + O_T2, 169);
  convt_kernel<<<ew(8 * 784), 256, 0, stream>>>(ws + O_T2, ws + O_M2, 13, 13, 4, 2, 28, 28);
  t_mul_kernel<<<ew(8 * 8 * 784), 256, 0, stream>>>(ws + O_V1, ws + O_M2, ws + O_T1, 784);
  convt_kernel<<<ew(8 * 900), 256, 0, stream>>>(ws + O_T1, ws + O_M1, 28, 28, 3, 1, 30, 30);
  map_caps_kernel<<<ew(8 * 16 * 900), 256, 0, stream>>>(ws + O_P, ws + O_M1, out + 16);
}
